// SE3FormerBlocks_91156385890603
// MI455X (gfx1250) — matrix operand order repaired and checked by reading, not compiled, not run on hardware
//
#include <hip/hip_runtime.h>
#include <hip/hip_bf16.h>
#include <math.h>

#define L_N   4096
#define C_N   128
#define H_N   8
#define NB_N  2
#define NL_N  4
#define DSEQ_N 640
#define NRBF_N 30

typedef __attribute__((ext_vector_type(16))) _Float16 v16h;
typedef __attribute__((ext_vector_type(8)))  float    v8f;

// ---------- order-preserving float<->uint for atomic max ----------
__device__ __forceinline__ unsigned enc_f(float f) {
  unsigned u = __float_as_uint(f);
  return (u & 0x80000000u) ? ~u : (u | 0x80000000u);
}
__device__ __forceinline__ float dec_f(unsigned u) {
  return (u & 0x80000000u) ? __uint_as_float(u & 0x7fffffffu) : __uint_as_float(~u);
}

// =================================================================
// Wave32 WMMA GEMM:  C[M,N] (+)= Arows[M,K] @ W[K,N]
//   - A row r: if rowidx, r = rowidx[m] else r = m (first source only)
//   - K split: k < K1 from A (lda=K1), k >= K1 from A2 (lda=K-K1, row=m)
//   - 128 threads = 4 waves; wave tile 16x64 (4 f32 accumulators);
//     block tile 64x64. B tile staged f32->f16 into LDS in [n][k]
//     layout so each lane's 16-half fragment is one contiguous 32B
//     LDS read. One A fragment feeds 4 WMMAs per K-step.
// =================================================================
__global__ void wmma_gemm_kernel(const float* __restrict__ A,
                                 const float* __restrict__ A2,
                                 const int*   __restrict__ rowidx,
                                 int K1,
                                 const float* __restrict__ W,
                                 float* __restrict__ Cmat,
                                 int M, int K, int N, int acc)
{
  const int lane = threadIdx.x & 31;
  const int wave = threadIdx.x >> 5;          // 0..3
  const int mbase = blockIdx.x * 64 + wave * 16;
  const int nblk  = blockIdx.y * 64;

  __shared__ __align__(32) _Float16 Bs[64 * 32];   // [n][k], 4 KB

  int mrow = mbase + (lane & 15);
  if (mrow >= M) mrow = M - 1;
  const int r = rowidx ? rowidx[mrow] : mrow;

  // C/D layout: lane<16 -> N=lane, M=+0..7 ; lane>=16 -> N=lane-16, M=+8..15
  const int crow0 = mbase + ((lane >> 4) << 3);
  const int ccol0 = nblk + (lane & 15);

  v8f c[4] = {{}, {}, {}, {}};
  if (acc) {
#pragma unroll
    for (int j = 0; j < 4; ++j) {
      const int cc = ccol0 + (j << 4);
#pragma unroll
      for (int i = 0; i < 8; ++i) {
        int rr = crow0 + i;
        c[j][i] = (rr < M && cc < N) ? Cmat[(size_t)rr * N + cc] : 0.f;
      }
    }
  }

  for (int k0 = 0; k0 < K; k0 += 32) {
    // stage B tile (32 k x 64 n) into LDS, transposed to [n][k].
    // Global side: consecutive threads read consecutive n (coalesced).
    for (int t = threadIdx.x; t < 64 * 32; t += 128) {
      int kk = t >> 6, nn = t & 63;
      int n = nblk + nn;
      Bs[nn * 32 + kk] = (_Float16)((n < N) ? W[(size_t)(k0 + kk) * N + n] : 0.f);
    }
    __syncthreads();

    // A fragment (16x32 f16): lane<16: halves 0..15=K+0..15
    //                         lane>=16: halves 0..15=K+16..31
    const float* arow;
    int klocal;
    if (k0 < K1) { arow = A  + (size_t)r    * K1;       klocal = k0; }
    else         { arow = A2 + (size_t)mrow * (K - K1); klocal = k0 - K1; }
    const int kb = klocal + ((lane >> 4) << 4);

    const float4 a0 = *(const float4*)(arow + kb);
    const float4 a1 = *(const float4*)(arow + kb + 4);
    const float4 a2 = *(const float4*)(arow + kb + 8);
    const float4 a3 = *(const float4*)(arow + kb + 12);
    v16h a;
    a[0]  = (_Float16)a0.x; a[1]  = (_Float16)a0.y;
    a[2]  = (_Float16)a0.z; a[3]  = (_Float16)a0.w;
    a[4]  = (_Float16)a1.x; a[5]  = (_Float16)a1.y;
    a[6]  = (_Float16)a1.z; a[7]  = (_Float16)a1.w;
    a[8]  = (_Float16)a2.x; a[9]  = (_Float16)a2.y;
    a[10] = (_Float16)a2.z; a[11] = (_Float16)a2.w;
    a[12] = (_Float16)a3.x; a[13] = (_Float16)a3.y;
    a[14] = (_Float16)a3.z; a[15] = (_Float16)a3.w;

    // B fragment: lanes 0-15 hold K=k0+0..15, lanes 16-31 hold K=k0+16..31.
    // [n][k] LDS layout -> one contiguous 32B read per fragment.
    const int kb2 = (lane >> 4) << 4;      // 0 or 16
    const int nn0 = lane & 15;
#pragma unroll
    for (int j = 0; j < 4; ++j) {
      const v16h b = *(const v16h*)(Bs + (nn0 + (j << 4)) * 32 + kb2);
      c[j] = __builtin_amdgcn_wmma_f32_16x16x32_f16(false, a, false, b,
                                                    (short)0, c[j], false, false);
    }
    __syncthreads();
  }

#pragma unroll
  for (int j = 0; j < 4; ++j) {
    const int cc = ccol0 + (j << 4);
#pragma unroll
    for (int i = 0; i < 8; ++i) {
      int rr = crow0 + i;
      if (rr < M && cc < N) Cmat[(size_t)rr * N + cc] = c[j][i];
    }
  }
}

// =================================================================
// Small kernels
// =================================================================
__global__ void build_edges_kernel(const int* __restrict__ src0, const int* __restrict__ dst0,
                                   int* __restrict__ srcdir, int* __restrict__ dstdir, int E0)
{
  int e = blockIdx.x * blockDim.x + threadIdx.x;
  if (e >= E0) return;
  srcdir[e]      = src0[e]; dstdir[e]      = dst0[e];
  srcdir[e + E0] = dst0[e]; dstdir[e + E0] = src0[e];
}

__global__ void edge_feat_kernel(const float* __restrict__ xyz,
                                 const int* __restrict__ src0, const int* __restrict__ dst0,
                                 const float* __restrict__ etype, const float* __restrict__ pij,
                                 const float* __restrict__ mu, const float* __restrict__ sg,
                                 float* __restrict__ e0f, float* __restrict__ reln,
                                 float* __restrict__ e1b, int E0)
{
  int e = blockIdx.x * blockDim.x + threadIdx.x;
  if (e >= E0) return;
  int si = src0[e], di = dst0[e];
  float vx = xyz[di*3+0] - xyz[si*3+0];
  float vy = xyz[di*3+1] - xyz[si*3+1];
  float vz = xyz[di*3+2] - xyz[si*3+2];
  float d  = sqrtf(vx*vx + vy*vy + vz*vz);
  float* r0 = e0f + (size_t)e * 32;
  float* r1 = e0f + (size_t)(e + E0) * 32;
  r0[0] = r1[0] = etype[e];
  r0[1] = r1[1] = pij[e];
#pragma unroll
  for (int k = 0; k < NRBF_N; ++k) {
    float t = d - mu[k], s = sg[k];
    float f = __expf(-(t * t) / (2.f * s * s));
    r0[2 + k] = f; r1[2 + k] = f;
  }
  float inv = 1.f / (d + 1e-6f);
  reln[(size_t)e*3+0] =  vx*inv; reln[(size_t)e*3+1] =  vy*inv; reln[(size_t)e*3+2] =  vz*inv;
  reln[(size_t)(e+E0)*3+0] = -vx*inv; reln[(size_t)(e+E0)*3+1] = -vy*inv; reln[(size_t)(e+E0)*3+2] = -vz*inv;
  e1b[(size_t)e*3+0] = vx; e1b[(size_t)e*3+1] = vy; e1b[(size_t)e*3+2] = vz;
  e1b[(size_t)(e+E0)*3+0] = vx; e1b[(size_t)(e+E0)*3+1] = vy; e1b[(size_t)(e+E0)*3+2] = vz;
}

__global__ void v_init_kernel(const float* __restrict__ xyz, const float* __restrict__ w1,
                              float* __restrict__ vx, float* __restrict__ vy,
                              float* __restrict__ vz, int n)
{
  int idx = blockIdx.x * blockDim.x + threadIdx.x;
  if (idx >= n * C_N) return;
  int l = idx >> 7, c = idx & 127;
  float w = w1[c];
  vx[idx] = xyz[l*3+0] * w;
  vy[idx] = xyz[l*3+1] * w;
  vz[idx] = xyz[l*3+2] * w;
}

__global__ void ln_kernel(const float* __restrict__ s, float* __restrict__ sn, int n)
{
  int row  = blockIdx.x * 4 + (threadIdx.x >> 5);
  int lane = threadIdx.x & 31;
  if (row >= n) return;
  const float* x = s + (size_t)row * C_N;
  float v0 = x[lane], v1 = x[lane+32], v2 = x[lane+64], v3 = x[lane+96];
  float sum = v0 + v1 + v2 + v3;
  for (int o = 16; o; o >>= 1) sum += __shfl_xor(sum, o, 32);
  float m = sum * (1.f / 128.f);
  float d0 = v0 - m, d1 = v1 - m, d2 = v2 - m, d3 = v3 - m;
  float vs = d0*d0 + d1*d1 + d2*d2 + d3*d3;
  for (int o = 16; o; o >>= 1) vs += __shfl_xor(vs, o, 32);
  float inv = rsqrtf(vs * (1.f / 128.f) + 1e-5f);
  float* y = sn + (size_t)row * C_N;
  y[lane] = d0*inv; y[lane+32] = d1*inv; y[lane+64] = d2*inv; y[lane+96] = d3*inv;
}

__global__ void logits_kernel(const float* __restrict__ q, const float* __restrict__ kbuf,
                              const int* __restrict__ dstdir,
                              float* __restrict__ logits, unsigned* __restrict__ mxenc,
                              int cs, int cn)
{
  int idx = blockIdx.x * blockDim.x + threadIdx.x;
  if (idx >= cn * H_N) return;
  int el = idx >> 3, h = idx & 7;
  int e = cs + el;
  int d = dstdir[e];
  const float* qp = q    + (size_t)d  * C_N + h * 16;
  const float* kp = kbuf + (size_t)el * C_N + h * 16;
  float acc = 0.f;
#pragma unroll
  for (int j = 0; j < 16; ++j) acc += qp[j] * kp[j];
  float lg = acc * 0.25f;  // 1/sqrt(DK=16)
  logits[(size_t)e * H_N + h] = lg;
  atomicMax(&mxenc[d * H_N + h], enc_f(lg));
}

__global__ void ex_kernel(const float* __restrict__ logits, const int* __restrict__ dstdir,
                          const unsigned* __restrict__ mxenc, float* __restrict__ den,
                          float* __restrict__ exb, int E)
{
  int idx = blockIdx.x * blockDim.x + threadIdx.x;
  if (idx >= E * H_N) return;
  int e = idx >> 3, h = idx & 7;
  int d = dstdir[e];
  unsigned me = mxenc[d * H_N + h];
  float mx = me ? dec_f(me) : 0.f;  // 0 sentinel == no finite max -> 0 (ref semantics)
  float ex = __expf(logits[idx] - mx);
  exb[idx] = ex;
  atomicAdd(&den[d * H_N + h], ex);
}

__global__ void aggregate_kernel(const float* __restrict__ v0b, const float* __restrict__ grb,
                                 const float* __restrict__ g1b,
                                 const float* __restrict__ v1xb, const float* __restrict__ v1yb,
                                 const float* __restrict__ v1zb,
                                 const float* __restrict__ exb, const int* __restrict__ dstdir,
                                 const float* __restrict__ reln, const float* __restrict__ e1b,
                                 float* __restrict__ agg0, float* __restrict__ a1x,
                                 float* __restrict__ a1y, float* __restrict__ a1z,
                                 int cs, int cn)
{
  long idx = (long)blockIdx.x * blockDim.x + threadIdx.x;
  if (idx >= (long)cn * C_N) return;
  int el = (int)(idx >> 7), hd = (int)(idx & 127), h = hd >> 4;
  int e = cs + el;
  int d = dstdir[e];
  float a  = exb[(size_t)e * H_N + h];
  float gr = grb[idx], g1 = g1b[idx];
  float rx = reln[(size_t)e*3+0], ry = reln[(size_t)e*3+1], rz = reln[(size_t)e*3+2];
  float ex1 = e1b[(size_t)e*3+0], ey1 = e1b[(size_t)e*3+1], ez1 = e1b[(size_t)e*3+2];
  size_t o = (size_t)d * C_N + hd;
  atomicAdd(&agg0[o], a * v0b[idx]);
  atomicAdd(&a1x[o],  a * (v1xb[idx] + gr*rx + g1*ex1));
  atomicAdd(&a1y[o],  a * (v1yb[idx] + gr*ry + g1*ey1));
  atomicAdd(&a1z[o],  a * (v1zb[idx] + gr*rz + g1*ez1));
}

__global__ void normalize_kernel(float* __restrict__ agg0, float* __restrict__ a1x,
                                 float* __restrict__ a1y, float* __restrict__ a1z,
                                 const float* __restrict__ den, int n)
{
  int idx = blockIdx.x * blockDim.x + threadIdx.x;
  if (idx >= n * C_N) return;
  int node = idx >> 7, h = (idx >> 4) & 7;
  float sc = 1.f / (den[node * H_N + h] + 1e-9f);
  agg0[idx] *= sc; a1x[idx] *= sc; a1y[idx] *= sc; a1z[idx] *= sc;
}

__global__ void xyz_update_kernel(const float* __restrict__ vx, const float* __restrict__ vy,
                                  const float* __restrict__ vz, const float* __restrict__ wout,
                                  float* __restrict__ xyz, int n)
{
  int node = blockIdx.x * 4 + (threadIdx.x >> 5);
  int lane = threadIdx.x & 31;
  if (node >= n) return;
  float dx = 0.f, dy = 0.f, dz = 0.f;
  for (int c = lane; c < C_N; c += 32) {
    float w = wout[c];
    size_t o = (size_t)node * C_N + c;
    dx += vx[o] * w; dy += vy[o] * w; dz += vz[o] * w;
  }
  for (int o = 16; o; o >>= 1) {
    dx += __shfl_xor(dx, o, 32); dy += __shfl_xor(dy, o, 32); dz += __shfl_xor(dz, o, 32);
  }
  if (lane == 0) {
    xyz[node*3+0] += dx; xyz[node*3+1] += dy; xyz[node*3+2] += dz;
  }
}

// =================================================================
// Host orchestration
// =================================================================
extern "C" void kernel_launch(void* const* d_in, const int* in_sizes, int n_in,
                              void* d_out, int out_size, void* d_ws, size_t ws_size,
                              hipStream_t stream)
{
  const float* seq_rep  = (const float*)d_in[0];
  const float* xyz_init = (const float*)d_in[1];
  const float* rbf_mu   = (const float*)d_in[2];
  const float* rbf_sg   = (const float*)d_in[3];
  const float* etype    = (const float*)d_in[4];
  const float* pij      = (const float*)d_in[5];

  // params dict may be flattened in insertion order or jax sorted-key order.
  const float *Win0,*Win1,*Wq,*Wk,*Wv0,*Wv1,*Wrel,*We1,*Wo0,*Wo1,*Wout1;
  if (in_sizes[6] == 32768) {
    // sorted: We1, Win0, Win1, Wk, Wo0, Wo1, Wout1, Wq, Wrel, Wv0, Wv1
    We1  = (const float*)d_in[6];  Win0 = (const float*)d_in[7];
    Win1 = (const float*)d_in[8];  Wk   = (const float*)d_in[9];
    Wo0  = (const float*)d_in[10]; Wo1  = (const float*)d_in[11];
    Wout1= (const float*)d_in[12]; Wq   = (const float*)d_in[13];
    Wrel = (const float*)d_in[14]; Wv0  = (const float*)d_in[15];
    Wv1  = (const float*)d_in[16];
  } else {
    // insertion: Win0, Win1, Wq, Wk, Wv0, Wv1, Wrel, We1, Wo0, Wo1, Wout1
    Win0 = (const float*)d_in[6];  Win1 = (const float*)d_in[7];
    Wq   = (const float*)d_in[8];  Wk   = (const float*)d_in[9];
    Wv0  = (const float*)d_in[10]; Wv1  = (const float*)d_in[11];
    Wrel = (const float*)d_in[12]; We1  = (const float*)d_in[13];
    Wo0  = (const float*)d_in[14]; Wo1  = (const float*)d_in[15];
    Wout1= (const float*)d_in[16];
  }
  const int* esrc = (const int*)d_in[17];
  const int* edst = (const int*)d_in[18];

  const int E0 = in_sizes[4];
  const int E  = 2 * E0;

  // ---- workspace carve-up ----
  char* wp = (char*)d_ws;
  auto alloc = [&](size_t nbytes) -> void* {
    void* p = (void*)wp; wp += (nbytes + 255) & ~(size_t)255; return p;
  };
  const size_t LC = (size_t)L_N * C_N * sizeof(float);
  float* s    = (float*)alloc(LC);
  float* sn   = (float*)alloc(LC);
  float* q    = (float*)alloc(LC);
  float* vx   = (float*)alloc(LC);
  float* vy   = (float*)alloc(LC);
  float* vz   = (float*)alloc(LC);
  float* agg0 = (float*)alloc(LC);
  float* a1x  = (float*)alloc(LC);
  float* a1y  = (float*)alloc(LC);
  float* a1z  = (float*)alloc(LC);
  float*    den   = (float*)   alloc((size_t)L_N * H_N * sizeof(float));
  unsigned* mxenc = (unsigned*)alloc((size_t)L_N * H_N * sizeof(unsigned));
  int* srcdir = (int*)alloc((size_t)E * sizeof(int));
  int* dstdir = (int*)alloc((size_t)E * sizeof(int));
  float* e0f  = (float*)alloc((size_t)E * 32 * sizeof(float));
  float* reln = (float*)alloc((size_t)E * 3  * sizeof(float));
  float* e1b  = (float*)alloc((size_t)E * 3  * sizeof(float));
  float* logits = (float*)alloc((size_t)E * H_N * sizeof(float));
  float* exb    = (float*)alloc((size_t)E * H_N * sizeof(float));

  size_t used = (size_t)(wp - (char*)d_ws);
  size_t remain = ws_size > used ? ws_size - used : 0;
  long chunk = (long)(remain / (6ull * C_N * sizeof(float)));
  if (chunk > 32768) chunk = 32768;
  chunk &= ~63L;
  if (chunk < 64) chunk = 64;
  const size_t PB = (size_t)chunk * C_N * sizeof(float);
  float* p0 = (float*)alloc(PB);   // kbuf / v0buf
  float* p1 = (float*)alloc(PB);   // grel
  float* p2 = (float*)alloc(PB);   // ge1
  float* p3 = (float*)alloc(PB);   // v1x
  float* p4 = (float*)alloc(PB);   // v1y
  float* p5 = (float*)alloc(PB);   // v1z

  float* xyz = (float*)d_out;      // xyz lives in the output buffer
  hipMemcpyAsync(xyz, xyz_init, (size_t)L_N * 3 * sizeof(float),
                 hipMemcpyDeviceToDevice, stream);

  build_edges_kernel<<<(E0 + 255) / 256, 256, 0, stream>>>(esrc, edst, srcdir, dstdir, E0);

  auto gemm = [&](const float* A, const float* A2, const int* ridx, int K1,
                  const float* W, float* Cm, int M, int K, int N, int acc) {
    dim3 g((unsigned)((M + 63) / 64), (unsigned)((N + 63) / 64), 1);
    wmma_gemm_kernel<<<g, 128, 0, stream>>>(A, A2, ridx, K1, W, Cm, M, K, N, acc);
  };

  for (int b = 0; b < NB_N; ++b) {
    // edge geometry from current xyz (recomputed per block, like the reference)
    edge_feat_kernel<<<(E0 + 127) / 128, 128, 0, stream>>>(
        xyz, esrc, edst, etype, pij, rbf_mu, rbf_sg, e0f, reln, e1b, E0);

    // fiber_in projections
    gemm(seq_rep, nullptr, nullptr, DSEQ_N, Win0 + (size_t)b * DSEQ_N * C_N,
         s, L_N, DSEQ_N, C_N, 0);
    v_init_kernel<<<(L_N * C_N + 255) / 256, 256, 0, stream>>>(
        xyz, Win1 + (size_t)b * C_N, vx, vy, vz, L_N);

    for (int l = 0; l < NL_N; ++l) {
      const size_t wi = (size_t)b * NL_N + l;
      const float* Wq_  = Wq   + wi * C_N * C_N;
      const float* Wk_  = Wk   + wi * (C_N + 32) * C_N;
      const float* Wv0_ = Wv0  + wi * (C_N + 32) * C_N;
      const float* Wv1_ = Wv1  + wi * C_N * C_N;
      const float* Wrel_= Wrel + wi * 32 * C_N;
      const float* We1_ = We1  + wi * 32 * C_N;
      const float* Wo0_ = Wo0  + wi * C_N * C_N;
      const float* Wo1_ = Wo1  + wi * C_N * C_N;

      ln_kernel<<<L_N / 4, 128, 0, stream>>>(s, sn, L_N);
      gemm(sn, nullptr, nullptr, C_N, Wq_, q, L_N, C_N, C_N, 0);

      hipMemsetAsync(mxenc, 0, (size_t)L_N * H_N * sizeof(unsigned), stream);
      hipMemsetAsync(den,   0, (size_t)L_N * H_N * sizeof(float),    stream);

      // pass 1: k = [sn[src]|e0] @ Wk, logits + running max
      for (int cs = 0; cs < E; cs += (int)chunk) {
        int cn = (E - cs < chunk) ? (E - cs) : (int)chunk;
        gemm(sn, e0f + (size_t)cs * 32, srcdir + cs, C_N, Wk_, p0, cn, C_N + 32, C_N, 0);
        logits_kernel<<<(cn * H_N + 255) / 256, 256, 0, stream>>>(
            q, p0, dstdir, logits, mxenc, cs, cn);
      }
      ex_kernel<<<(E * H_N + 255) / 256, 256, 0, stream>>>(
          logits, dstdir, mxenc, den, exb, E);

      hipMemsetAsync(agg0, 0, LC, stream);
      hipMemsetAsync(a1x,  0, LC, stream);
      hipMemsetAsync(a1y,  0, LC, stream);
      hipMemsetAsync(a1z,  0, LC, stream);

      // pass 2: messages + scatter-add aggregation
      for (int cs = 0; cs < E; cs += (int)chunk) {
        int cn = (E - cs < chunk) ? (E - cs) : (int)chunk;
        gemm(sn, e0f + (size_t)cs * 32, srcdir + cs, C_N, Wv0_, p0, cn, C_N + 32, C_N, 0);
        gemm(e0f + (size_t)cs * 32, nullptr, nullptr, 32, Wrel_, p1, cn, 32, C_N, 0);
        gemm(e0f + (size_t)cs * 32, nullptr, nullptr, 32, We1_,  p2, cn, 32, C_N, 0);
        gemm(vx, nullptr, srcdir + cs, C_N, Wv1_, p3, cn, C_N, C_N, 0);
        gemm(vy, nullptr, srcdir + cs, C_N, Wv1_, p4, cn, C_N, C_N, 0);
        gemm(vz, nullptr, srcdir + cs, C_N, Wv1_, p5, cn, C_N, C_N, 0);
        long nthr = (long)cn * C_N;
        aggregate_kernel<<<(unsigned)((nthr + 255) / 256), 256, 0, stream>>>(
            p0, p1, p2, p3, p4, p5, exb, dstdir, reln, e1b,
            agg0, a1x, a1y, a1z, cs, cn);
      }

      normalize_kernel<<<(L_N * C_N + 255) / 256, 256, 0, stream>>>(
          agg0, a1x, a1y, a1z, den, L_N);

      // residual updates (accumulate GEMMs)
      gemm(agg0, nullptr, nullptr, C_N, Wo0_, s,  L_N, C_N, C_N, 1);
      gemm(a1x,  nullptr, nullptr, C_N, Wo1_, vx, L_N, C_N, C_N, 1);
      gemm(a1y,  nullptr, nullptr, C_N, Wo1_, vy, L_N, C_N, C_N, 1);
      gemm(a1z,  nullptr, nullptr, C_N, Wo1_, vz, L_N, C_N, C_N, 1);
    }

    // fiber_out {1:1}: xyz += v . Wout1
    xyz_update_kernel<<<L_N / 4, 128, 0, stream>>>(
        vx, vy, vz, Wout1 + (size_t)b * C_N, xyz, L_N);
  }
  (void)n_in; (void)out_size;
}
